// LSTM_Actor_71906342470219
// MI455X (gfx1250) — matrix operand order repaired and checked by reading, not compiled, not run on hardware
//
#include <hip/hip_runtime.h>
#include <hip/hip_bf16.h>

// ---- CDNA5 WMMA fragment types (wave32) ----
typedef __attribute__((ext_vector_type(16))) __bf16 v16bf;
typedef __attribute__((ext_vector_type(8)))  __bf16 v8bf;
typedef __attribute__((ext_vector_type(8)))  float  v8f;

#define T_STEPS 512
#define BATCH   2048
#define DIM     64
#define HID     64
#define ADIM    16

__device__ __forceinline__ float fsig(float x) {
    // sigmoid via v_exp_f32 + v_rcp_f32 (no slow IEEE divide)
    return __builtin_amdgcn_rcpf(1.0f + __expf(-x));
}
__device__ __forceinline__ float ftanh(float x) {
    // tanh(x) = 1 - 2/(exp(2x)+1); saturates correctly at +/-1
    return 1.0f - 2.0f * __builtin_amdgcn_rcpf(__expf(2.0f * x) + 1.0f);
}

__device__ __forceinline__ v8f wmma_bf16(v16bf a, v16bf b, v8f c) {
    return __builtin_amdgcn_wmma_f32_16x16x32_bf16(
        /*neg_a=*/false, a, /*neg_b=*/false, b,
        /*c_mod=*/(short)0, c, /*reuse_a=*/false, /*reuse_b=*/false);
}

// Block = 16 batch rows, 8 waves.
//   Waves 0-3 ("recurrent"): wave cg owns hidden cols [16cg,16cg+16) of all 4
//     gates; per step: gates = xw_lds + h@W_hh^T (8 WMMAs), LSTM cell, publish
//     h (bf16) to LDS, round-robin output projection.
//   Waves 4-7 ("input"): wave cg computes xw(t+1) = bias + x_{t+1}@W_ih^T for
//     the same gate tiles (8 WMMAs) and publishes f32 accumulators to LDS,
//     one step ahead (pure GEMM, off the recurrence critical path).
__global__ __launch_bounds__(256, 1)
void lstm_actor_kernel(const float* __restrict__ x,
                       const float* __restrict__ W_ih,
                       const float* __restrict__ W_hh,
                       const float* __restrict__ b_ih,
                       const float* __restrict__ b_hh,
                       const float* __restrict__ W_out,
                       const float* __restrict__ b_out,
                       float* __restrict__ out)
{
    // h tile: 16 rows x 64 cols bf16, row stride 80 halfs (160B -> 16B-aligned)
    __shared__ __align__(16) __bf16 hbuf[2][16][80];
    // xw hand-off: [buf][col-group][gate][lane*8] f32, raw C/D-layout 32B/lane
    __shared__ __align__(32) float xwbuf[2][4][4][32 * 8];

    const int tid   = threadIdx.x;
    const int lane  = tid & 31;
    const int wv    = __builtin_amdgcn_readfirstlane(tid >> 5); // uniform 0..7
    const bool isRec = (wv < 4);
    const int cg    = wv & 3;                 // column group: hidden cols 16*cg..
    const int n16   = lane & 15;              // N within 16-wide tile
    const int off   = (lane < 16) ? 0 : 8;    // A-frag K offset == C/D M-half offset
    const int koffB = (lane < 16) ? 0 : 16;   // unused; B-frags use off (0/8)
    const int bBase = blockIdx.x * 16;        // batch tile origin
    const int m     = n16;                    // A-frag row (batch row in tile)

    // ---- role-specific main weight fragments share ONE register array ----
    // B[k][n] = W[n][k]; per lane a B-fragment = k {off..off+7, off+16..off+23} of row n.
    const float* Wsrc = isRec ? W_hh : W_ih;  // both [256 x 64], uniform select
    v16bf Wf[4][2];
    #pragma unroll
    for (int q = 0; q < 4; ++q) {
        const int n = q * 64 + cg * 16 + n16;  // gate column (i,f,g,o blocks)
        #pragma unroll
        for (int kf = 0; kf < 2; ++kf) {
            const float* wp = Wsrc + n * DIM + kf * 32 + off;
            #pragma unroll
            for (int e = 0; e < 16; ++e) Wf[q][kf][e] = (__bf16)wp[e < 8 ? e : e + 8];
        }
    }
    v16bf Bout[2];
    #pragma unroll
    for (int kf = 0; kf < 2; ++kf) {
        const float* wo = W_out + n16 * HID + kf * 32 + off;
        #pragma unroll
        for (int e = 0; e < 16; ++e) Bout[kf][e] = (__bf16)wo[e < 8 ? e : e + 8];
    }
    float biasv[4];
    #pragma unroll
    for (int q = 0; q < 4; ++q) {
        const int n = q * 64 + cg * 16 + n16;
        biasv[q] = b_ih[n] + b_hh[n];          // fused bias (input waves use it)
    }
    const float bo = b_out[n16];

    // ---- producer step: xw(tt) = bias + x_tt @ W_ih^T -> xwbuf[tt&1] ----
    auto xw_step = [&](int tt) {
        v16bf xf[2];
        const float* xrow = x + ((size_t)tt * BATCH + bBase + m) * DIM;
        #pragma unroll
        for (int kf = 0; kf < 2; ++kf) {
            const float* p = xrow + kf * 32 + off;
            const float4 a0 = *(const float4*)(p);
            const float4 a1 = *(const float4*)(p + 4);
            const float4 a2 = *(const float4*)(p + 16);
            const float4 a3 = *(const float4*)(p + 20);
            const float tmp[16] = {a0.x, a0.y, a0.z, a0.w, a1.x, a1.y, a1.z, a1.w,
                                   a2.x, a2.y, a2.z, a2.w, a3.x, a3.y, a3.z, a3.w};
            #pragma unroll
            for (int e = 0; e < 16; ++e) xf[kf][e] = (__bf16)tmp[e];
        }
        float* xws = &xwbuf[tt & 1][cg][0][0];
        #pragma unroll
        for (int q = 0; q < 4; ++q) {
            v8f acc;
            #pragma unroll
            for (int i = 0; i < 8; ++i) acc[i] = biasv[q];
            acc = wmma_bf16(xf[0], Wf[q][0], acc);
            acc = wmma_bf16(xf[1], Wf[q][1], acc);
            *(v8f*)(xws + q * 256 + lane * 8) = acc;
        }
    };

    // ---- recurrent state ----
    v16bf hf[2];                               // h as bf16 A-fragments
    #pragma unroll
    for (int e = 0; e < 16; ++e) { hf[0][e] = (__bf16)0.0f; hf[1][e] = (__bf16)0.0f; }
    v8f c_st;
    #pragma unroll
    for (int i = 0; i < 8; ++i) c_st[i] = 0.0f;

    // Prologue: producers fill xw(0); everyone syncs.
    if (!isRec) xw_step(0);
    __syncthreads();

    for (int t = 0; t < T_STEPS; ++t) {
        if (isRec) {
            // ---- gates = xw(t) + h@W_hh^T  (8 WMMAs on the critical path) ----
            const float* xws = &xwbuf[t & 1][cg][0][0];
            v8f g[4];
            #pragma unroll
            for (int q = 0; q < 4; ++q) {
                v8f acc = *(const v8f*)(xws + q * 256 + lane * 8);
                acc = wmma_bf16(hf[0], Wf[q][0], acc);
                acc = wmma_bf16(hf[1], Wf[q][1], acc);
                g[q] = acc;
            }
            // ---- LSTM cell (fp32; C/D layout is gate-aligned) ----
            v8f hnew;
            #pragma unroll
            for (int i = 0; i < 8; ++i) {
                const float ig = fsig(g[0][i]);
                const float fg = fsig(g[1][i]);
                const float gg = ftanh(g[2][i]);
                const float og = fsig(g[3][i]);
                const float c  = fg * c_st[i] + ig * gg;
                c_st[i] = c;
                hnew[i] = og * ftanh(c);
            }
            // ---- publish h_new (bf16) to ping-pong LDS tile ----
            const int hcol = cg * 16 + n16;
            #pragma unroll
            for (int r = 0; r < 8; ++r)
                hbuf[t & 1][r + off][hcol] = (__bf16)hnew[r];
        } else if (t + 1 < T_STEPS) {
            // ---- producers run one step ahead ----
            xw_step(t + 1);
        }

        __syncthreads();

        if (isRec) {
            // ---- reload full h(t) as A-fragments (ds_load_b128 x4) ----
            #pragma unroll
            for (int kf = 0; kf < 2; ++kf) {
                const __bf16* hr = &hbuf[t & 1][m][kf * 32 + off];
                const v8bf lo = *(const v8bf*)(hr);
                const v8bf hi = *(const v8bf*)(hr + 16);
                #pragma unroll
                for (int e = 0; e < 8; ++e) { hf[kf][e] = lo[e]; hf[kf][e + 8] = hi[e]; }
            }
            // ---- output projection, round-robined over recurrent waves ----
            if ((t & 3) == cg) {
                v8f oacc;
                #pragma unroll
                for (int i = 0; i < 8; ++i) oacc[i] = bo;
                oacc = wmma_bf16(hf[0], Bout[0], oacc);
                oacc = wmma_bf16(hf[1], Bout[1], oacc);
                float* orow = out + ((size_t)t * BATCH + bBase) * ADIM + n16;
                #pragma unroll
                for (int r = 0; r < 8; ++r)
                    orow[(size_t)(r + off) * ADIM] = oacc[r];
            }
        }
    }
}

extern "C" void kernel_launch(void* const* d_in, const int* in_sizes, int n_in,
                              void* d_out, int out_size, void* d_ws, size_t ws_size,
                              hipStream_t stream) {
    (void)in_sizes; (void)n_in; (void)out_size; (void)d_ws; (void)ws_size;
    const float* x     = (const float*)d_in[0];
    const float* W_ih  = (const float*)d_in[1];
    const float* W_hh  = (const float*)d_in[2];
    const float* b_ih  = (const float*)d_in[3];
    const float* b_hh  = (const float*)d_in[4];
    const float* W_out = (const float*)d_in[5];
    const float* b_out = (const float*)d_in[6];
    float* out = (float*)d_out;

    dim3 grid(BATCH / 16);   // 128 batch tiles
    dim3 block(256);         // 8 waves: 4 recurrent + 4 input producers
    lstm_actor_kernel<<<grid, block, 0, stream>>>(x, W_ih, W_hh, b_ih, b_hh,
                                                  W_out, b_out, out);
}
